// EdgeRegressionHead_63333587746832
// MI455X (gfx1250) — matrix operand order repaired and checked by reading, not compiled, not run on hardware
//
#include <hip/hip_runtime.h>
#include <hip/hip_bf16.h>

// CDNA5 / gfx1250: wave32, WMMA 16x16x32 f16 -> f32
typedef _Float16 half_t;
typedef __attribute__((ext_vector_type(16))) _Float16 v16h;
typedef __attribute__((ext_vector_type(8)))  _Float16 v8h;
typedef __attribute__((ext_vector_type(8)))  float    v8f;

#define DDIM    128          // hidden dim
#define W1_K    256          // 2*D
#define TILE_M  16           // edges per wave-tile
#define NWAVES  8            // waves per block (256 threads)
#define TILES_PER_WAVE 4     // amortize weight staging over 4 tiles per wave
#define LDS_PAD 8            // keep rows 16B aligned, spread banks
#define W1_LD   (W1_K + LDS_PAD)   // 264 halves per row
#define W2_LD   (DDIM + LDS_PAD)   // 136
#define ST_LD   (DDIM + LDS_PAD)   // 136

__global__ __launch_bounds__(256) void edge_mlp_wmma_kernel(
    const float* __restrict__ x,
    const int*   __restrict__ src,
    const int*   __restrict__ dst,
    const float* __restrict__ W1, const float* __restrict__ b1,
    const float* __restrict__ W2, const float* __restrict__ b2,
    const float* __restrict__ W3, const float* __restrict__ b3,
    float* __restrict__ out, int E)
{
    // ~136 KB LDS: weights transposed to [n][k] so B-fragments are contiguous.
    __shared__ half_t sW1[DDIM][W1_LD];                 // 66 KB
    __shared__ half_t sW2[DDIM][W2_LD];                 // 34 KB
    __shared__ half_t sStage[NWAVES][TILE_M][ST_LD];    // 34 KB (per-wave h1/h2 relayout)
    __shared__ float  sB1[DDIM];
    __shared__ float  sB2[DDIM];
    __shared__ float  sW3[DDIM];

    const int tid = threadIdx.x;

    // ---- stage weights (fp32 -> f16, transposed), once per block ----
    for (int i = tid; i < W1_K * DDIM; i += 256) {
        int k = i / DDIM, n = i % DDIM;       // W1 is [256][128] row-major
        sW1[n][k] = (half_t)W1[i];
    }
    for (int i = tid; i < DDIM * DDIM; i += 256) {
        int k = i / DDIM, n = i % DDIM;       // W2 is [128][128] row-major
        sW2[n][k] = (half_t)W2[i];
    }
    for (int i = tid; i < DDIM; i += 256) {
        sB1[i] = b1[i]; sB2[i] = b2[i]; sW3[i] = W3[i];
    }
    __syncthreads();

    const int wave = tid >> 5;
    const int lane = tid & 31;
    const int mrow = lane & 15;   // M index (edge within tile) / N index for B frags
    const int khi  = lane >> 4;   // lane-half selects K stripe
    const float b3v = b3[0];

    half_t (*stage)[ST_LD] = sStage[wave];

    const long waveBase =
        ((long)blockIdx.x * NWAVES + wave) * (long)(TILES_PER_WAVE * TILE_M);

    #pragma unroll 1
    for (int t = 0; t < TILES_PER_WAVE; ++t) {
        const long tileBase = waveBase + (long)t * TILE_M;
        long eRow = tileBase + mrow;
        if (eRow > (long)E - 1) eRow = (long)E - 1;   // clamp tail (store is guarded)
        const float* rowS = x + (size_t)src[eRow] * DDIM;
        const float* rowD = x + (size_t)dst[eRow] * DDIM;

        v8f acc[8];
        #pragma unroll
        for (int nt = 0; nt < 8; ++nt)
            #pragma unroll
            for (int r = 0; r < 8; ++r) acc[nt][r] = 0.0f;

        // ================= Layer 1: [16,256] @ [256,128] =================
        // A layout (16-bit 16x32): lane m = L%16; slots j<8 -> K = kt*32 + khi*8 + j,
        //                          slots j>=8 -> K = kt*32 + 16 + khi*8 + (j-8)
        #pragma unroll
        for (int kt = 0; kt < 8; ++kt) {
            const float* rp = (kt < 4) ? rowS : rowD;  // concat(x[src], x[dst])
            const int base = (kt & 3) * 32 + khi * 8;
            float4 f0 = *(const float4*)(rp + base);
            float4 f1 = *(const float4*)(rp + base + 4);
            float4 f2 = *(const float4*)(rp + base + 16);
            float4 f3 = *(const float4*)(rp + base + 20);
            v16h a;
            a[0]=(half_t)f0.x;  a[1]=(half_t)f0.y;  a[2]=(half_t)f0.z;  a[3]=(half_t)f0.w;
            a[4]=(half_t)f1.x;  a[5]=(half_t)f1.y;  a[6]=(half_t)f1.z;  a[7]=(half_t)f1.w;
            a[8]=(half_t)f2.x;  a[9]=(half_t)f2.y;  a[10]=(half_t)f2.z; a[11]=(half_t)f2.w;
            a[12]=(half_t)f3.x; a[13]=(half_t)f3.y; a[14]=(half_t)f3.z; a[15]=(half_t)f3.w;

            #pragma unroll
            for (int nt = 0; nt < 8; ++nt) {
                // B layout (32x16): lane n = L%16; j<8 -> K = khi*8 + j, j>=8 -> K = 16 + khi*8 + (j-8)
                const half_t* wp = &sW1[nt * 16 + mrow][kt * 32 + khi * 8];
                v8h blo = *(const v8h*)(wp);
                v8h bhi = *(const v8h*)(wp + 16);
                v16h b;
                #pragma unroll
                for (int j = 0; j < 8; ++j) { b[j] = blo[j]; b[8 + j] = bhi[j]; }
                acc[nt] = __builtin_amdgcn_wmma_f32_16x16x32_f16(
                    false, a, false, b, (short)0, acc[nt], false, false);
            }
        }

        // bias + ReLU, relayout D (lane=N, vgpr=M) -> stage[m][n] as f16
        #pragma unroll
        for (int nt = 0; nt < 8; ++nt) {
            float bias = sB1[nt * 16 + mrow];
            #pragma unroll
            for (int r = 0; r < 8; ++r) {
                float v = acc[nt][r] + bias;
                v = v > 0.0f ? v : 0.0f;
                stage[r + khi * 8][nt * 16 + mrow] = (half_t)v;
                acc[nt][r] = 0.0f;     // recycle accumulators for layer 2
            }
        }
        // Same-wave LDS ops are in-order; stage is private to this wave -> no barrier.

        // ================= Layer 2: [16,128] @ [128,128] =================
        #pragma unroll
        for (int kt = 0; kt < 4; ++kt) {
            const half_t* srow = &stage[mrow][0];
            const int kb = kt * 32 + khi * 8;
            v8h alo = *(const v8h*)(srow + kb);
            v8h ahi = *(const v8h*)(srow + kb + 16);
            v16h a;
            #pragma unroll
            for (int j = 0; j < 8; ++j) { a[j] = alo[j]; a[8 + j] = ahi[j]; }

            #pragma unroll
            for (int nt = 0; nt < 8; ++nt) {
                const half_t* wp = &sW2[nt * 16 + mrow][kt * 32 + khi * 8];
                v8h blo = *(const v8h*)(wp);
                v8h bhi = *(const v8h*)(wp + 16);
                v16h b;
                #pragma unroll
                for (int j = 0; j < 8; ++j) { b[j] = blo[j]; b[8 + j] = bhi[j]; }
                acc[nt] = __builtin_amdgcn_wmma_f32_16x16x32_f16(
                    false, a, false, b, (short)0, acc[nt], false, false);
            }
        }

        // bias + ReLU, write h2 back into stage (cols 0..127)
        #pragma unroll
        for (int nt = 0; nt < 8; ++nt) {
            float bias = sB2[nt * 16 + mrow];
            #pragma unroll
            for (int r = 0; r < 8; ++r) {
                float v = acc[nt][r] + bias;
                v = v > 0.0f ? v : 0.0f;
                stage[r + khi * 8][nt * 16 + mrow] = (half_t)v;
            }
        }

        // ================= Layer 3: [16,128] @ [128,1] (VALU dot) =================
        float s = 0.0f;
        {
            const half_t* hrow = &stage[mrow][0];
            const int kb3 = khi * 64;     // lane-halves split the 128-K dot
            #pragma unroll
            for (int k = 0; k < 64; ++k)
                s += (float)hrow[kb3 + k] * sW3[kb3 + k];
        }
        s += __shfl_xor(s, 16, 32);       // combine the two K-halves (wave32)

        if (lane < 16) {
            long e = tileBase + lane;
            if (e < (long)E) out[e] = s + b3v;
        }
    }
}

extern "C" void kernel_launch(void* const* d_in, const int* in_sizes, int n_in,
                              void* d_out, int out_size, void* d_ws, size_t ws_size,
                              hipStream_t stream) {
    const float* x  = (const float*)d_in[0];
    const int*   s  = (const int*)  d_in[1];
    const int*   d  = (const int*)  d_in[2];
    const float* W1 = (const float*)d_in[3];
    const float* b1 = (const float*)d_in[4];
    const float* W2 = (const float*)d_in[5];
    const float* b2 = (const float*)d_in[6];
    const float* W3 = (const float*)d_in[7];
    const float* b3 = (const float*)d_in[8];
    float* out = (float*)d_out;

    const int E = in_sizes[1];
    const int tiles  = (E + TILE_M - 1) / TILE_M;
    const int tilesPerBlock = NWAVES * TILES_PER_WAVE;
    const int blocks = (tiles + tilesPerBlock - 1) / tilesPerBlock;
    edge_mlp_wmma_kernel<<<blocks, 256, 0, stream>>>(x, s, d, W1, b1, W2, b2, W3, b3, out, E);
}
